// SeismicLSTM_37684043055120
// MI455X (gfx1250) — matrix operand order repaired and checked by reading, not compiled, not run on hardware
//
#include <hip/hip_runtime.h>
#include <stdint.h>

// ---- CDNA5 (gfx1250) types for WMMA ----
typedef __bf16    bf16;
typedef bf16      bf16x16 __attribute__((ext_vector_type(16)));
typedef float     f32x8   __attribute__((ext_vector_type(8)));
typedef float     f32x4   __attribute__((ext_vector_type(4)));
typedef uint32_t  u32;
typedef uint32_t  u32x4   __attribute__((ext_vector_type(4)));
typedef uint32_t  u32x8   __attribute__((ext_vector_type(8)));

#define TSEQ 512
#define HDIM 128
#define GDIM 512   // 4*H gate width
#define MB   16    // batch rows per workgroup (one WMMA M-tile)

// Branch-free native activations: one v_exp_f32 + one v_rcp_f32 each.
__device__ __forceinline__ float fsig(float x) {
    return __builtin_amdgcn_rcpf(1.0f + __builtin_amdgcn_exp2f(-1.44269504f * x));
}
__device__ __forceinline__ float ftanh(float x) {
    // tanh(x) = 2*sigmoid(2x) - 1
    return fmaf(2.0f, __builtin_amdgcn_rcpf(1.0f + __builtin_amdgcn_exp2f(-2.88539008f * x)), -1.0f);
}

// Load a 16(M)x32(K) bf16 A-fragment from LDS (row-major [16][128]).
// ISA wave32 layout: lane<16 holds row=lane, K {0..7,16..23}; lane>=16 holds
// row=lane-16, K {8..15,24..31}. Each run of 8 bf16 = one 16B contiguous read.
__device__ __forceinline__ bf16x16 loadA(const bf16* sH, int lane, int kt) {
    const int row  = lane & 15;
    const int half = lane >> 4;
    const bf16* p = sH + row * HDIM + kt * 32 + half * 8;
    const u32x4 lo = *(const u32x4*)(p);
    const u32x4 hi = *(const u32x4*)(p + 16);
    union { u32x8 u; bf16x16 v; } c;
    c.u[0] = lo[0]; c.u[1] = lo[1]; c.u[2] = lo[2]; c.u[3] = lo[3];
    c.u[4] = hi[0]; c.u[5] = hi[1]; c.u[6] = hi[2]; c.u[7] = hi[3];
    return c.v;
}

// Build a 32(K)x16(N) bf16 B-fragment in registers from a row-major fp32
// weight matrix W[N][K] (K = 128). ISA layout: lane holds column n = lane%16;
// lanes 0-15 carry K {0..7,16..23} of the 32-K tile, lanes 16-31 carry K {8..15,24..31}.
__device__ __forceinline__ bf16x16 loadB(const float* __restrict__ W, int lane, int ncol, int kt) {
    const int khalf = lane >> 4;
    const float* p = W + ncol * HDIM + kt * 32 + khalf * 8;
    bf16x16 b;
#pragma unroll
    for (int e = 0; e < 16; ++e) b[e] = (bf16)p[e + (e >= 8 ? 8 : 0)];
    return b;
}

// Pre-swizzled B-fragment spill area in LDS: [slot][half][lane] x 16B.
// 16B lane stride -> consecutive lanes hit distinct bank quads (full spread
// over 16 lanes, 2-way conflict between half-waves, the b128 minimum).
__device__ __forceinline__ void storeFragLDS(u32* sWF, int slot, int lane, bf16x16 v) {
    union { bf16x16 v; u32x8 u; } c; c.v = v;
    u32x4 lo, hi;
    lo[0] = c.u[0]; lo[1] = c.u[1]; lo[2] = c.u[2]; lo[3] = c.u[3];
    hi[0] = c.u[4]; hi[1] = c.u[5]; hi[2] = c.u[6]; hi[3] = c.u[7];
    *(u32x4*)(sWF + ((slot * 2 + 0) * 32 + lane) * 4) = lo;
    *(u32x4*)(sWF + ((slot * 2 + 1) * 32 + lane) * 4) = hi;
}
__device__ __forceinline__ bf16x16 loadFragLDS(const u32* sWF, int slot, int lane) {
    const u32x4 lo = *(const u32x4*)(sWF + ((slot * 2 + 0) * 32 + lane) * 4);
    const u32x4 hi = *(const u32x4*)(sWF + ((slot * 2 + 1) * 32 + lane) * 4);
    union { u32x8 u; bf16x16 v; } c;
    c.u[0] = lo[0]; c.u[1] = lo[1]; c.u[2] = lo[2]; c.u[3] = lo[3];
    c.u[4] = hi[0]; c.u[5] = hi[1]; c.u[6] = hi[2]; c.u[7] = hi[3];
    return c.v;
}

__global__ __launch_bounds__(512, 1)
void lstm2_fused_head(const float* __restrict__ x,
                      const float* __restrict__ Wih0, const float* __restrict__ Whh0,
                      const float* __restrict__ bih0, const float* __restrict__ bhh0,
                      const float* __restrict__ Wih1, const float* __restrict__ Whh1,
                      const float* __restrict__ bih1, const float* __restrict__ bhh1,
                      const float* __restrict__ W1,   const float* __restrict__ b1,
                      const float* __restrict__ W2,   const float* __restrict__ b2,
                      float* __restrict__ out)
{
    extern __shared__ char smem[];
    float* sG  = (float*)(smem);               // [16][512] raw gates (32KB), reused by head
    float* sXT = (float*)(smem + 32 * 1024);   // [512][16] input tile, transposed (32KB)
    bf16*  sH0 = (bf16*)(smem + 64 * 1024);    // [16][128] layer-0 hidden (4KB)
    bf16*  sH1 = (bf16*)(smem + 68 * 1024);    // [16][128] layer-1 hidden (4KB)
    u32*   sWF = (u32*)(smem + 72 * 1024);     // W_ih1 fragments, pre-swizzled (128KB)

    const int tid  = threadIdx.x;
    const int lane = tid & 31;
    const int wv   = tid >> 5;                 // wave 0..15, owns gate cols [32w, 32w+32)
    const int m0   = blockIdx.x * MB;

    // Stage this tile's input sequence transposed: sXT[t][m] = x[m0+m][t].
#pragma unroll 4
    for (int i = tid; i < MB * TSEQ; i += 512) {
        const int m = i >> 9;                  // i / TSEQ
        const int t = i & (TSEQ - 1);
        sXT[t * MB + m] = x[m0 * TSEQ + i];
    }

    // Zero hidden state.
    for (int i = tid; i < MB * HDIM; i += 512) { sH0[i] = (bf16)0.0f; sH1[i] = (bf16)0.0f; }

    // Per-lane gate-column constants (two N-tiles per wave -> two columns per lane).
    const int nA = wv * 32 + (lane & 15);
    const int nB = nA + 16;
    const float wg0A = Wih0[nA],            wg0B = Wih0[nB];             // IN == 1
    const float bs0A = bih0[nA] + bhh0[nA], bs0B = bih0[nB] + bhh0[nB];
    const float bs1A = bih1[nA] + bhh1[nA], bs1B = bih1[nB] + bhh1[nB];

    // Recurrent weights register-resident (128 VGPRs); W_ih1 fragments go to
    // LDS pre-swizzled so the inner loop reloads them with 2x ds_load_b128.
    bf16x16 fHH0[2][4], fHH1[2][4];
#pragma unroll
    for (int nt = 0; nt < 2; ++nt) {
        const int nc = wv * 32 + nt * 16 + (lane & 15);
#pragma unroll
        for (int kt = 0; kt < 4; ++kt) {
            fHH0[nt][kt] = loadB(Whh0, lane, nc, kt);
            fHH1[nt][kt] = loadB(Whh1, lane, nc, kt);
            storeFragLDS(sWF, wv * 8 + nt * 4 + kt, lane, loadB(Wih1, lane, nc, kt));
        }
    }

    // fp32 cell state: thread handles (m,j) elements tid, tid+512, tid+1024, tid+1536.
    float c0r[4] = {0.f, 0.f, 0.f, 0.f};
    float c1r[4] = {0.f, 0.f, 0.f, 0.f};

    __syncthreads();

    const int hiHalf = lane >> 4;  // accumulator layout: VGPR r -> M = r + 8*hiHalf
    const int ncol   = wv * 32 + (lane & 15);

    for (int t = 0; t < TSEQ; ++t) {
        // x values for this lane's 8 accumulator rows: contiguous in sXT[t][..].
        const f32x4 xlo = *(const f32x4*)(sXT + t * MB + 8 * hiHalf);
        const f32x4 xhi = *(const f32x4*)(sXT + t * MB + 8 * hiHalf + 4);
        float xv[8];
#pragma unroll
        for (int r = 0; r < 4; ++r) { xv[r] = xlo[r]; xv[r + 4] = xhi[r]; }

        // ---------------- layer 0: gates = x*Wih0 + b + h0 @ Whh0^T ----------------
        f32x8 a0, a1;
#pragma unroll
        for (int r = 0; r < 8; ++r) { a0[r] = fmaf(xv[r], wg0A, bs0A); a1[r] = fmaf(xv[r], wg0B, bs0B); }
#pragma unroll
        for (int kt = 0; kt < 4; ++kt) {
            bf16x16 af = loadA(sH0, lane, kt);
            a0 = __builtin_amdgcn_wmma_f32_16x16x32_bf16(false, af, false, fHH0[0][kt], (short)0, a0, false, false);
            a1 = __builtin_amdgcn_wmma_f32_16x16x32_bf16(false, af, false, fHH0[1][kt], (short)0, a1, false, false);
        }
#pragma unroll
        for (int r = 0; r < 8; ++r) {
            const int mrow = r + 8 * hiHalf;
            sG[mrow * GDIM + ncol]      = a0[r];
            sG[mrow * GDIM + ncol + 16] = a1[r];
        }
        __syncthreads();

        // layer-0 pointwise cell update (cross-wave gather of i/f/g/o via LDS).
#pragma unroll
        for (int q = 0; q < 4; ++q) {
            const int e = tid + q * 512;         // 0..2047 over 16x128
            const int m = e >> 7, j = e & 127;
            const float ig = fsig(sG[m * GDIM + j]);
            const float fg = fsig(sG[m * GDIM + j + 128]);
            const float gg = ftanh(sG[m * GDIM + j + 256]);
            const float og = fsig(sG[m * GDIM + j + 384]);
            const float c  = fg * c0r[q] + ig * gg;
            c0r[q] = c;
            sH0[m * HDIM + j] = (bf16)(og * ftanh(c));
        }
        __syncthreads();

        // ---- layer 1: two independent 4-deep WMMA chains (IH1 and HH1), added in f32 ----
        f32x8 p0, p1, q0, q1;
#pragma unroll
        for (int r = 0; r < 8; ++r) { p0[r] = bs1A; p1[r] = bs1B; q0[r] = 0.f; q1[r] = 0.f; }
#pragma unroll
        for (int kt = 0; kt < 4; ++kt) {
            bf16x16 af = loadA(sH0, lane, kt);
            bf16x16 w0 = loadFragLDS(sWF, wv * 8 + 0 * 4 + kt, lane);
            bf16x16 w1 = loadFragLDS(sWF, wv * 8 + 1 * 4 + kt, lane);
            p0 = __builtin_amdgcn_wmma_f32_16x16x32_bf16(false, af, false, w0, (short)0, p0, false, false);
            p1 = __builtin_amdgcn_wmma_f32_16x16x32_bf16(false, af, false, w1, (short)0, p1, false, false);
        }
#pragma unroll
        for (int kt = 0; kt < 4; ++kt) {
            bf16x16 bf = loadA(sH1, lane, kt);
            q0 = __builtin_amdgcn_wmma_f32_16x16x32_bf16(false, bf, false, fHH1[0][kt], (short)0, q0, false, false);
            q1 = __builtin_amdgcn_wmma_f32_16x16x32_bf16(false, bf, false, fHH1[1][kt], (short)0, q1, false, false);
        }
#pragma unroll
        for (int r = 0; r < 8; ++r) {
            const int mrow = r + 8 * hiHalf;
            sG[mrow * GDIM + ncol]      = p0[r] + q0[r];
            sG[mrow * GDIM + ncol + 16] = p1[r] + q1[r];
        }
        __syncthreads();

#pragma unroll
        for (int q = 0; q < 4; ++q) {
            const int e = tid + q * 512;
            const int m = e >> 7, j = e & 127;
            const float ig = fsig(sG[m * GDIM + j]);
            const float fg = fsig(sG[m * GDIM + j + 128]);
            const float gg = ftanh(sG[m * GDIM + j + 256]);
            const float og = fsig(sG[m * GDIM + j + 384]);
            const float c  = fg * c1r[q] + ig * gg;
            c1r[q] = c;
            sH1[m * HDIM + j] = (bf16)(og * ftanh(c));
        }
        __syncthreads();
    }

    // ---------------- head: relu(h1 @ W1^T + b1) @ W2^T + b2 ----------------
#pragma unroll
    for (int q = 0; q < 2; ++q) {
        const int e = tid + q * 512;             // 0..1023 over 16x64
        const int m = e >> 6, u = e & 63;
        float acc = b1[u];
        const float* w   = W1 + u * HDIM;
        const bf16* hrow = sH1 + m * HDIM;
#pragma unroll 8
        for (int k = 0; k < HDIM; ++k) acc += (float)hrow[k] * w[k];
        sG[m * 64 + u] = fmaxf(acc, 0.0f);       // reuse sG as [16][64] hidden
    }
    __syncthreads();

    if (tid < MB * 8) {
        const int m = tid >> 3, cc = tid & 7;
        float acc = b2[cc];
        const float* w    = W2 + cc * 64;
        const float* hrow = sG + m * 64;
#pragma unroll
        for (int u = 0; u < 64; ++u) acc += hrow[u] * w[u];
        out[(m0 + m) * 8 + cc] = acc;
    }
}

extern "C" void kernel_launch(void* const* d_in, const int* in_sizes, int n_in,
                              void* d_out, int out_size, void* d_ws, size_t ws_size,
                              hipStream_t stream) {
    (void)in_sizes; (void)n_in; (void)out_size; (void)d_ws; (void)ws_size;
    const float* x    = (const float*)d_in[0];
    const float* Wih0 = (const float*)d_in[1];
    const float* Whh0 = (const float*)d_in[2];
    const float* bih0 = (const float*)d_in[3];
    const float* bhh0 = (const float*)d_in[4];
    const float* Wih1 = (const float*)d_in[5];
    const float* Whh1 = (const float*)d_in[6];
    const float* bih1 = (const float*)d_in[7];
    const float* bhh1 = (const float*)d_in[8];
    const float* W1   = (const float*)d_in[9];
    const float* b1   = (const float*)d_in[10];
    const float* W2   = (const float*)d_in[11];
    const float* b2   = (const float*)d_in[12];
    float* out = (float*)d_out;

    const size_t shmem = 200 * 1024;  // sG 32K + sXT 32K + sH0 4K + sH1 4K + sWF 128K
    // Opt in to >64KB dynamic LDS (host-side, capture-safe; ignore errors).
    (void)hipFuncSetAttribute((const void*)lstm2_fused_head,
                              hipFuncAttributeMaxDynamicSharedMemorySize, (int)shmem);

    const int nblocks = 2048 / MB;    // 128 workgroups, 16 batch rows each
    lstm2_fused_head<<<dim3(nblocks), dim3(512), shmem, stream>>>(
        x, Wih0, Whh0, bih0, bhh0, Wih1, Whh1, bih1, bhh1, W1, b1, W2, b2, out);
}
